// DeformableTemporalAttention_47167330845294
// MI455X (gfx1250) — matrix operand order repaired and checked by reading, not compiled, not run on hardware
//
#include <hip/hip_runtime.h>
#include <math.h>

// Problem constants from the reference setup (compile-time; B derived from sizes).
#define HH 8          // NUM_HEADS
#define PP 9          // NUM_POINTS
#define TT 3
#define DD 256
#define HP 64
#define WPX 64
#define NN (HP * WPX) // 4096

typedef __attribute__((ext_vector_type(16))) __bf16 v16bf;
typedef __attribute__((ext_vector_type(8)))  float  v8f;

static __device__ __forceinline__ int imin(int a, int b) { return a < b ? a : b; }

#if defined(__has_builtin)
#if __has_builtin(__builtin_amdgcn_sched_barrier)
#define SCHED_FENCE() __builtin_amdgcn_sched_barrier(0)
#endif
#endif
#ifndef SCHED_FENCE
#define SCHED_FENCE()
#endif

// ---------------------------------------------------------------------------
// WMMA fragment layouts (CDNA5 ISA 7.12.2, wave32):
//   A 16x32 bf16 : lane L row M=L%16; element e=0..7 <-> K=kbA+0..7,
//                  e=8..15 <-> K=16+kbA+0..7, kbA = 0 / 8 (lane<16 / >=16)
//   B 32x16 bf16 : lane L col N=L%16; e=0..7 <-> K=kbB+0..7, e=8..15 <-> K=16+kbB+0..7, kbB = 0 / 8
//   C/D 8 f32    : VGPR g -> row M = g + (lane<16 ? 0 : 8), col N = lane%16
// ---------------------------------------------------------------------------

// ---------------------------------------------------------------------------
// Weight pre-pack: W (f32, row-major 256 x Ncols) -> bf16 in B-fragment order.
// Entry (ct, kt, lane) = 16 contiguous bf16 (32 B). Out-of-range cols -> 0.
// ---------------------------------------------------------------------------
__global__ void pack_w_bf16(const float* __restrict__ W, int ldw, int Ncols,
                            __bf16* __restrict__ pW, int nct) {
  const int tid = blockIdx.x * blockDim.x + threadIdx.x;
  if (tid >= nct * 256) return;                 // nct * 8 kt * 32 lanes
  const int lane = tid & 31;
  const int kt   = (tid >> 5) & 7;
  const int ct   = tid >> 8;
  const int c    = ct * 16 + (lane & 15);
  const int kb   = (lane < 16) ? 0 : 8;
  __bf16* dst = pW + (long)tid * 16;
#pragma unroll
  for (int e = 0; e < 16; ++e) {
    const int k = kt * 32 + kb + (e < 8 ? e : e + 8);
    dst[e] = (c < Ncols) ? (__bf16)W[(long)k * ldw + c] : (__bf16)0.f;
  }
}

// ---------------------------------------------------------------------------
// Cooperative A-tile stage: 16 rows x 256 f32 -> bf16 A-fragment order in LDS.
// 256 threads; thread t reads 16 consecutive f32 (coalesced 64B) and scatters
// two 16B chunks into fragment entries. LDS = 8 kt * 32 lanes * 16 bf16 = 8KB.
// ---------------------------------------------------------------------------
static __device__ __forceinline__ void stage_a_lds(__bf16* __restrict__ lA,
                                                   const float* __restrict__ A,
                                                   long lda, int row0, int tid) {
  const int row = tid >> 4;                 // 0..15
  const int k0  = (tid & 15) * 16;          // 0..240, multiple of 16
  const float* ap = A + (long)(row0 + row) * lda + k0;
  const float4 q0 = *(const float4*)(ap + 0);
  const float4 q1 = *(const float4*)(ap + 4);
  const float4 q2 = *(const float4*)(ap + 8);
  const float4 q3 = *(const float4*)(ap + 12);
  const int kt   = k0 >> 5;
  const int half = (k0 & 16) ? 8 : 0;       // element offset within entry
  // k0..k0+7  -> lane-lo entry (row),     elements half..half+7
  __bf16* d0 = lA + ((kt * 32 + row) * 16) + half;
  d0[0]=(__bf16)q0.x; d0[1]=(__bf16)q0.y; d0[2]=(__bf16)q0.z; d0[3]=(__bf16)q0.w;
  d0[4]=(__bf16)q1.x; d0[5]=(__bf16)q1.y; d0[6]=(__bf16)q1.z; d0[7]=(__bf16)q1.w;
  // k0+8..k0+15 -> lane-hi entry (row+16), elements half..half+7
  __bf16* d1 = lA + ((kt * 32 + row + 16) * 16) + half;
  d1[0]=(__bf16)q2.x; d1[1]=(__bf16)q2.y; d1[2]=(__bf16)q2.z; d1[3]=(__bf16)q2.w;
  d1[4]=(__bf16)q3.x; d1[5]=(__bf16)q3.y; d1[6]=(__bf16)q3.z; d1[7]=(__bf16)q3.w;
}

// Full K=256 accumulation for one 16x16 tile.
// B: all 8 fragments (16 x b128) issued as one in-flight batch from global.
// A: streamed from LDS with a 1-deep pipeline -- ds_load of frag kt+1 sits in
// the same sched region as the WMMA on frag kt, so only 2 A-frags are live.
// Peak pressure ~95 VGPRs -> no scratch spills; partial loadcnt waits drain B.
static __device__ __forceinline__ v8f tile_accum_packed(const __bf16* __restrict__ lA,
                                                        const __bf16* __restrict__ pW,
                                                        int ct, int lane) {
  const __bf16* p = pW + ((long)ct * 256 + lane) * 16;
  v16bf bf[8];
#pragma unroll
  for (int kt = 0; kt < 8; ++kt)
    bf[kt] = *(const v16bf*)(p + kt * 512);           // 32 lanes * 16 elems
  SCHED_FENCE();                                      // keep B batch hoisted

  v16bf ac = *(const v16bf*)(lA + (long)lane * 16);   // A frag kt = 0
  v8f acc = {};
#pragma unroll
  for (int kt = 0; kt < 8; ++kt) {
    v16bf an = ac;
    if (kt < 7) an = *(const v16bf*)(lA + ((kt + 1) * 32 + lane) * 16);
    acc = __builtin_amdgcn_wmma_f32_16x16x32_bf16(false, ac, false, bf[kt],
                                                  (short)0, acc, false, false);
    SCHED_FENCE();                                    // pin 1-deep A pipeline
    ac = an;
  }
  return acc;
}

// ---------------------------------------------------------------------------
// Generic GEMM: C[M x Ncols] = A[M x 256] * W + bias, row-major C.
// Block = 8 waves, owns one 16-row tile (LDS-staged); waves stride col tiles.
// M must be a multiple of 16 (true for all launches) -> no row masking.
// ---------------------------------------------------------------------------
__global__ void wmma_gemm_bias(const float* __restrict__ A, long lda,
                               const __bf16* __restrict__ pW,
                               const float* __restrict__ bias,
                               float* __restrict__ C, long ldc,
                               int Ncols) {
  __shared__ __bf16 lA[8 * 32 * 16];
  const int tid  = threadIdx.x;
  const int lane = tid & 31;
  const int wave = tid >> 5;
  const int row0 = blockIdx.x * 16;

  stage_a_lds(lA, A, lda, row0, tid);
  __syncthreads();

  const int nct   = (Ncols + 15) >> 4;
  const int rbase = row0 + ((lane < 16) ? 0 : 8);
  for (int ct = wave; ct < nct; ct += 8) {
    // branch-free speculative prefetch of the next packed tile (clamped)
    __builtin_prefetch(pW + ((long)imin(ct + 8, nct - 1) * 256 + lane) * 16, 0, 1);

    const v8f acc = tile_accum_packed(lA, pW, ct, lane);
    const int col = ct * 16 + (lane & 15);
    if (col < Ncols) {                       // single exec region for all 8 stores
      const float bv = bias[col];
      float* cp = C + (long)rbase * ldc + col;
#pragma unroll
      for (int g = 0; g < 8; ++g) cp[(long)g * ldc] = acc[g] + bv;
    }
  }
}

// ---------------------------------------------------------------------------
// Value GEMM: value = x @ W_v + b_v, scattered into sampler layout
// val[(b*HH+h)*TT + t][n][c]   (channels innermost -> coalesced corner gathers)
// Ncols = 256 exactly; M multiple of 16 -> no masking at all.
// ---------------------------------------------------------------------------
__global__ void wmma_value_gemm(const float* __restrict__ x,
                                const __bf16* __restrict__ pWv,
                                const float* __restrict__ bv,
                                float* __restrict__ val) {
  __shared__ __bf16 lA[8 * 32 * 16];
  const int tid  = threadIdx.x;
  const int lane = tid & 31;
  const int wave = tid >> 5;
  const int row0 = blockIdx.x * 16;

  stage_a_lds(lA, x, DD, row0, tid);
  __syncthreads();

  const int rbase = row0 + ((lane < 16) ? 0 : 8);
  const int t0  = rbase % TT;                  // incremental (t, b*NN+n) walk
  const int nb0 = rbase / TT;
  for (int ct = wave; ct < 16; ct += 8) {      // 256 cols -> 16 tiles
    const v8f acc = tile_accum_packed(lA, pWv, ct, lane);
    const int col = ct * 16 + (lane & 15);
    const float bias = bv[col];
    const int h = col >> 5, c = col & 31;
    int t = t0, nb = nb0;
#pragma unroll
    for (int g = 0; g < 8; ++g) {
      const int n = nb & (NN - 1);             // NN = 4096 (power of two)
      const int b = nb >> 12;
      const long dst = ((((long)(b * HH + h) * TT + t) * NN) + n) * 32 + c;
      val[dst] = acc[g] + bias;
      if (++t == TT) { t = 0; ++nb; }
    }
  }
}

// ---------------------------------------------------------------------------
// Sampler: one wave per (b, n, h); lane = channel c in [0,32).
// softmax over 9 points, 27 bilinear samples (t x p), coalesced 128B gathers.
// ---------------------------------------------------------------------------
__global__ void deform_sample_kernel(const float* __restrict__ val,   // (B*HH*TT, NN, 32)
                                     const float* __restrict__ offl,  // (B*NN, 144)
                                     const float* __restrict__ attl,  // (B*NN, 72)
                                     float* __restrict__ agg)         // (B*NN, 256)
{
  const int gwave = (int)((blockIdx.x * blockDim.x + threadIdx.x) >> 5);
  const int lane  = threadIdx.x & 31;

  const int h  = gwave & (HH - 1);
  const int bn = gwave >> 3;            // b*NN + n
  const int n  = bn & (NN - 1);
  const int b  = bn >> 12;
  const int iy = n / WPX, ix = n % WPX;
  const float refx = (float)ix / (float)(WPX - 1) * 2.f - 1.f;
  const float refy = (float)iy / (float)(HP - 1) * 2.f - 1.f;

  // softmax over the 9 attention logits for this (b, n, h)
  const float* al = attl + (long)bn * (HH * PP) + h * PP;
  float w[PP], mx = -1e30f;
#pragma unroll
  for (int p = 0; p < PP; ++p) { w[p] = al[p]; mx = fmaxf(mx, w[p]); }
  float sum = 0.f;
#pragma unroll
  for (int p = 0; p < PP; ++p) { w[p] = __expf(w[p] - mx); sum += w[p]; }
  const float inv = 1.f / sum;

  const float* ol = offl + (long)bn * (HH * PP * 2) + h * (PP * 2);
  float acc = 0.f;
#pragma unroll
  for (int p = 0; p < PP; ++p) {
    const float gx = (refx + ol[2 * p + 0] + 1.f) * 0.5f * (float)(WPX - 1);
    const float gy = (refy + ol[2 * p + 1] + 1.f) * 0.5f * (float)(HP - 1);
    const float x0f = floorf(gx), y0f = floorf(gy);
    const int   x0 = (int)x0f, y0 = (int)y0f;
    const float wx1 = gx - x0f, wy1 = gy - y0f;
    const float wx0 = 1.f - wx1, wy0 = 1.f - wy1;
    float s = 0.f;
    for (int t = 0; t < TT; ++t) {
      const float* img = val + ((long)((b * HH + h) * TT + t)) * NN * 32;
      auto corner = [&](int xi, int yi, float wgt) -> float {
        if (xi < 0 || xi >= WPX || yi < 0 || yi >= HP) return 0.f;
        return img[((long)(yi * WPX + xi)) * 32 + lane] * wgt;
      };
      s += corner(x0,     y0,     wx0 * wy0);
      s += corner(x0 + 1, y0,     wx1 * wy0);
      s += corner(x0,     y0 + 1, wx0 * wy1);
      s += corner(x0 + 1, y0 + 1, wx1 * wy1);
    }
    acc += (w[p] * inv) * s;
  }
  agg[(long)bn * DD + h * 32 + lane] = acc;
}

// ---------------------------------------------------------------------------
extern "C" void kernel_launch(void* const* d_in, const int* in_sizes, int n_in,
                              void* d_out, int out_size, void* d_ws, size_t ws_size,
                              hipStream_t stream) {
  const float* x      = (const float*)d_in[0];
  const float* W_off  = (const float*)d_in[1];
  const float* b_off  = (const float*)d_in[2];
  const float* W_attn = (const float*)d_in[3];
  const float* b_attn = (const float*)d_in[4];
  const float* W_v    = (const float*)d_in[5];
  const float* b_v    = (const float*)d_in[6];
  const float* W_o    = (const float*)d_in[7];
  const float* b_o    = (const float*)d_in[8];

  const int B  = in_sizes[0] / (NN * TT * DD);   // = 4
  const int M1 = B * NN * TT;                    // value GEMM rows  = 49152
  const int M2 = B * NN;                         // query rows       = 16384

  // col-tile counts
  const int nctV = 16;                           // 256 cols
  const int nctO = (HH * PP * 2 + 15) / 16;      // 144 -> 9
  const int nctA = (HH * PP + 15) / 16;          // 72  -> 5
  const int nctF = 16;                           // 256 cols

  // workspace carve-up
  float* val  = (float*)d_ws;                              // B*HH*TT*NN*32  (48 MB)
  float* offl = val  + (long)B * HH * TT * NN * 32;        // B*NN*144
  float* attl = offl + (long)M2 * (HH * PP * 2);           // B*NN*72
  float* agg  = attl + (long)M2 * (HH * PP);               // B*NN*256
  __bf16* pWv   = (__bf16*)(agg + (long)M2 * DD);          // packed weights (bf16)
  __bf16* pWoff = pWv   + (long)nctV * 4096;
  __bf16* pWatt = pWoff + (long)nctO * 4096;
  __bf16* pWo   = pWatt + (long)nctA * 4096;

  // 0) pack weights into bf16 B-fragment order (tiny)
  pack_w_bf16<<<dim3(nctV), dim3(256), 0, stream>>>(W_v,   DD,          DD,          pWv,   nctV);
  pack_w_bf16<<<dim3(nctO), dim3(256), 0, stream>>>(W_off, HH * PP * 2, HH * PP * 2, pWoff, nctO);
  pack_w_bf16<<<dim3(nctA), dim3(256), 0, stream>>>(W_attn, HH * PP,    HH * PP,     pWatt, nctA);
  pack_w_bf16<<<dim3(nctF), dim3(256), 0, stream>>>(W_o,   DD,          DD,          pWo,   nctF);

  // 1) value = x @ W_v + b_v  -> sampler layout
  wmma_value_gemm<<<dim3(M1 / 16), dim3(256), 0, stream>>>(x, pWv, b_v, val);

  // 2) offset logits = query @ W_off + b_off  (query = x[:, :, T/2] -> base x+DD, lda=TT*DD)
  wmma_gemm_bias<<<dim3(M2 / 16), dim3(256), 0, stream>>>(
      x + DD, (long)TT * DD, pWoff, b_off, offl, HH * PP * 2, HH * PP * 2);

  // 3) attn logits = query @ W_attn + b_attn
  wmma_gemm_bias<<<dim3(M2 / 16), dim3(256), 0, stream>>>(
      x + DD, (long)TT * DD, pWatt, b_attn, attl, HH * PP, HH * PP);

  // 4) softmax + deformable bilinear sampling + head aggregation
  {
    const int waves  = B * NN * HH;              // 131072
    const int blocks = waves / 8;                // 8 waves / 256-thread block
    deform_sample_kernel<<<dim3(blocks), dim3(256), 0, stream>>>(val, offl, attl, agg);
  }

  // 5) out = agg @ W_o + b_o
  wmma_gemm_bias<<<dim3(M2 / 16), dim3(256), 0, stream>>>(
      agg, DD, pWo, b_o, (float*)d_out, DD, DD);
}
